// LocalGlobalCausalSelfAttention_21242908246342
// MI455X (gfx1250) — matrix operand order repaired and checked by reading, not compiled, not run on hardware
//
#include <hip/hip_runtime.h>

// MI455X / gfx1250, wave32. All contractions via v_wmma_f32_16x16x32_bf16.
// GEMM weight panels staged to LDS with global_load_async_to_lds_b128 (ASYNCcnt).
#define NHEAD 16
#define HDIM 64
#define WIN 256
#define BSTRIDE 40   // LDS row stride (ushorts) for 64x32 weight panel: 80B, bank-conflict-free

typedef __attribute__((ext_vector_type(16))) __bf16 v16bf;
typedef __attribute__((ext_vector_type(8)))  float  v8f;
typedef __attribute__((ext_vector_type(4)))  unsigned int v4u;

union Frag {
  v16bf v;
  unsigned short h[16];
  unsigned int   w[8];
  v4u            q[2];
};

__device__ __forceinline__ unsigned short f2bf(float f) {
  unsigned int u = __float_as_uint(f);
  u += 0x7fffu + ((u >> 16) & 1u);          // round-to-nearest-even
  return (unsigned short)(u >> 16);
}
__device__ __forceinline__ unsigned int packbf2(float a, float b) {
  return (unsigned int)f2bf(a) | ((unsigned int)f2bf(b) << 16);
}
__device__ __forceinline__ v8f wmma_bf16(const Frag& a, const Frag& b, v8f c) {
  return __builtin_amdgcn_wmma_f32_16x16x32_bf16(false, a.v, false, b.v,
                                                 (short)0, c, false, false);
}

// ---------------------------------------------------------------------------
// One-shot fp32 -> bf16 conversion (memory bound, outside all GEMM loops).
// ---------------------------------------------------------------------------
__global__ __launch_bounds__(256) void cvt_f32_bf16(
    const float* __restrict__ in, unsigned short* __restrict__ out, int n8) {
  int i = blockIdx.x * blockDim.x + threadIdx.x;
  if (i >= n8) return;
  const float* p = in + (size_t)i * 8;
  v4u r = {packbf2(p[0], p[1]), packbf2(p[2], p[3]),
           packbf2(p[4], p[5]), packbf2(p[6], p[7])};
  *(v4u*)(out + (size_t)i * 8) = r;
}

// ---------------------------------------------------------------------------
// V transpose: qkv[m, 2C + hd] -> Vt[(b*C + hd) * T + t]  (bf16)
// ---------------------------------------------------------------------------
__global__ __launch_bounds__(256) void vtrans_kernel(
    const unsigned short* __restrict__ qkv, unsigned short* __restrict__ vt,
    int B, int T) {
  const int C = NHEAD * HDIM;
  int e = blockIdx.x * blockDim.x + threadIdx.x;
  if (e >= B * T * C) return;
  int m = e / C, hd = e - m * C;
  int b = m / T, t = m - b * T;
  vt[((size_t)(b * C + hd)) * T + t] = qkv[(size_t)m * 3 * C + 2 * C + hd];
}

// ---------------------------------------------------------------------------
// GEMM (bf16 x bf16, f32 acc): Out[m,n] = sum_k A[m,k] * W[n,k]
// Block (8 waves) -> 256x64 tile. Per 32-wide k-step the block's shared
// 64x32 weight panel is DMA'd to LDS with global_load_async_to_lds_b128
// (one 16B chunk per thread), then consumed as ds_load_b128 B-fragments.
// Each wave computes a 32x64 sub-tile: 8 WMMA per k-step.
// ---------------------------------------------------------------------------
template <bool F32OUT>
__global__ __launch_bounds__(256) void gemm_bb(
    const unsigned short* __restrict__ A, const unsigned short* __restrict__ Wb,
    void* __restrict__ OutP, int M, int N, int K) {
  __shared__ unsigned short ldsB[64 * BSTRIDE];   // 5 KB staged weight panel

  int tid  = threadIdx.x;
  int wave = tid >> 5, lane = tid & 31;
  int l16 = lane & 15, half = lane >> 4;
  int ntn = N >> 6;
  int bm = blockIdx.x / ntn, bn = blockIdx.x - bm * ntn;
  int m0 = bm * 256 + wave * 32, n0 = bn << 6;

  // async-copy assignment: thread -> one 16B chunk of the 64x32 panel
  int brow = tid >> 2, bchunk = tid & 3;
  const unsigned short* bsrc = Wb + (size_t)(n0 + brow) * K + bchunk * 8;
  unsigned int blds = (unsigned int)(size_t)(ldsB + brow * BSTRIDE + bchunk * 8);

  v8f zero = {0.f,0.f,0.f,0.f,0.f,0.f,0.f,0.f};
  v8f acc[2][4] = {{zero, zero, zero, zero}, {zero, zero, zero, zero}};

  for (int k0 = 0; k0 < K; k0 += 32) {
    __syncthreads();   // previous panel fully consumed
    {
      unsigned long long gaddr = (unsigned long long)(bsrc + k0);
      asm volatile("global_load_async_to_lds_b128 %0, %1, off"
                   :: "v"(blds), "v"(gaddr) : "memory");
      asm volatile("s_wait_asynccnt 0x0" ::: "memory");
    }
    __syncthreads();   // whole panel visible to all waves

    Frag a[2];
#pragma unroll
    for (int mi = 0; mi < 2; ++mi) {
      const v4u* ap =
          (const v4u*)(A + (size_t)(m0 + mi*16 + l16) * K + k0 + half * 8);
      a[mi].q[0] = ap[0];
      a[mi].q[1] = ap[2];
    }
#pragma unroll
    for (int nf = 0; nf < 4; ++nf) {
      Frag bw;
      const v4u* bp = (const v4u*)(ldsB + (nf*16 + l16) * BSTRIDE + half * 8);
      bw.q[0] = bp[0];
      bw.q[1] = bp[2];
      acc[0][nf] = wmma_bf16(a[0], bw, acc[0][nf]);
      acc[1][nf] = wmma_bf16(a[1], bw, acc[1][nf]);
    }
  }

#pragma unroll
  for (int mi = 0; mi < 2; ++mi)
#pragma unroll
    for (int nf = 0; nf < 4; ++nf)
#pragma unroll
      for (int r = 0; r < 8; ++r) {
        size_t idx = (size_t)(m0 + mi*16 + half*8 + r) * N + n0 + nf*16 + l16;
        if (F32OUT) ((float*)OutP)[idx] = acc[mi][nf][r];
        else        ((unsigned short*)OutP)[idx] = f2bf(acc[mi][nf][r]);
      }
}

// ---------------------------------------------------------------------------
// Flash attention, transposed-S formulation:
//   S^T = K*Q^T  (query = lane -> per-lane scalar softmax state,
//                 reductions in-register + ONE shfl_xor(16))
//   O^T = V^T*P  (P B-frag built from registers via 8 packed half-swaps)
// One wave per (b, h, 16-query tile); 32 keys per iteration. No LDS.
// ---------------------------------------------------------------------------
__global__ __launch_bounds__(256) void attn_kernel(
    const unsigned short* __restrict__ QKV, const unsigned short* __restrict__ Vt,
    unsigned short* __restrict__ Yb, int Bb, int T) {
  const int C  = NHEAD * HDIM;     // 1024
  const int C3 = 3 * C;            // 3072

  int lane = threadIdx.x & 31;
  int l16 = lane & 15, half = lane >> 4;
  int tilesBH = T >> 4;
  int gw = blockIdx.x * (blockDim.x >> 5) + (threadIdx.x >> 5);
  int bh = gw / tilesBH;
  if (bh >= Bb * NHEAD) return;
  int q0 = (gw - bh * tilesBH) << 4;
  int b = bh / NHEAD, hh = bh - b * NHEAD;

  // Q^T B-fragments (col = query = l16; d = 32t + 8*half + s, +8 for s >= 8)
  Frag qb[2];
  const unsigned short* qrow = QKV + ((size_t)(b*T + q0 + l16)) * C3 + hh*HDIM;
#pragma unroll
  for (int t = 0; t < 2; ++t) {
    const v4u* qp = (const v4u*)(qrow + t*32 + half*8);
    qb[t].q[0] = qp[0];
    qb[t].q[1] = qp[2];
  }

  v8f zero = {0.f,0.f,0.f,0.f,0.f,0.f,0.f,0.f};
  v8f o[4] = {zero, zero, zero, zero};   // O^T tiles [16d x 16q]
  float m = -3.0e38f, l = 0.f;
  int iq  = q0 + l16;
  int ilo = iq - (WIN - 1);

  int jlo = q0 - (WIN - 1);
  int jstart = jlo > 0 ? (jlo & ~31) : 0;
  int jb = (jstart > 0) ? 0 : jstart;
  bool firstGlobal = (jstart > 0);

  while (jb <= q0 + 15) {
    // ---- S^T = K Q^T for 32 keys ----
    v8f s0 = zero, s1 = zero;
#pragma unroll
    for (int t = 0; t < 2; ++t) {
      Frag ka0, ka1;
      const v4u* kp0 = (const v4u*)(QKV + ((size_t)(b*T + jb +      l16))*C3 + C + hh*HDIM + t*32 + half*8);
      const v4u* kp1 = (const v4u*)(QKV + ((size_t)(b*T + jb + 16 + l16))*C3 + C + hh*HDIM + t*32 + half*8);
      ka0.q[0] = kp0[0]; ka0.q[1] = kp0[2];
      ka1.q[0] = kp1[0]; ka1.q[1] = kp1[2];
      s0 = wmma_bf16(ka0, qb[t], s0);
      s1 = wmma_bf16(ka1, qb[t], s1);
    }

    // ---- mask (wave-uniform fast path for interior blocks) ----
    float pv0[8], pv1[8];
    float mxo = -3.0e38f;
    if (jb + 31 <= q0 && jb >= q0 - (WIN - 16)) {
#pragma unroll
      for (int r = 0; r < 8; ++r) {
        pv0[r] = s0[r] * 0.125f;
        pv1[r] = s1[r] * 0.125f;
        mxo = fmaxf(mxo, fmaxf(pv0[r], pv1[r]));
      }
    } else {
#pragma unroll
      for (int r = 0; r < 8; ++r) {
        int j0 = jb + half*8 + r, j1 = j0 + 16;
        bool a0 = (j0 <= iq) && ((j0 >= ilo) || (j0 < 4));
        bool a1 = (j1 <= iq) && ((j1 >= ilo) || (j1 < 4));
        pv0[r] = a0 ? s0[r] * 0.125f : -3.0e38f;
        pv1[r] = a1 ? s1[r] * 0.125f : -3.0e38f;
        mxo = fmaxf(mxo, fmaxf(pv0[r], pv1[r]));
      }
    }
    mxo = fmaxf(mxo, __shfl_xor(mxo, 16, 32));
    float mn   = fmaxf(m, mxo);
    float corr = __expf(m - mn);
    m = mn;

    float rs = 0.f;
#pragma unroll
    for (int r = 0; r < 8; ++r) {
      pv0[r] = __expf(pv0[r] - mn);
      pv1[r] = __expf(pv1[r] - mn);
      rs += pv0[r] + pv1[r];
    }
    rs += __shfl_xor(rs, 16, 32);
    l = l * corr + rs;

    // ---- P B-fragment from registers (8 packed half-swaps) ----
    unsigned int own0[4], own1[4];
#pragma unroll
    for (int u = 0; u < 4; ++u) {
      own0[u] = packbf2(pv0[2*u], pv0[2*u+1]);
      own1[u] = packbf2(pv1[2*u], pv1[2*u+1]);
    }
    Frag pb;
#pragma unroll
    for (int u = 0; u < 4; ++u) {
      unsigned int o0 = (unsigned int)__shfl_xor((int)own0[u], 16, 32);
      unsigned int o1 = (unsigned int)__shfl_xor((int)own1[u], 16, 32);
      pb.w[u]     = half ? o1      : own0[u];
      pb.w[u + 4] = half ? own1[u] : o0;
    }

    // ---- O^T += V^T P ----
#pragma unroll
    for (int dt = 0; dt < 4; ++dt) {
      Frag va;
      const v4u* vp =
          (const v4u*)(Vt + ((size_t)(bh*HDIM + dt*16 + l16)) * T + jb + half*16);
      va.q[0] = vp[0];
      va.q[1] = vp[1];
      o[dt] *= corr;
      o[dt] = wmma_bf16(va, pb, o[dt]);
    }

    if (firstGlobal) { jb = jstart; firstGlobal = false; }
    else jb += 32;
  }

  float rinv = 1.0f / l;
  unsigned short* yrow = Yb + ((size_t)(b*T + q0 + l16)) * C + hh*HDIM;
#pragma unroll
  for (int dt = 0; dt < 4; ++dt)
#pragma unroll
    for (int u = 0; u < 4; ++u) {
      unsigned int w = packbf2(o[dt][2*u] * rinv, o[dt][2*u+1] * rinv);
      *(unsigned int*)(yrow + dt*16 + half*8 + 2*u) = w;
    }
}

// ---------------------------------------------------------------------------
extern "C" void kernel_launch(void* const* d_in, const int* in_sizes, int n_in,
                              void* d_out, int out_size, void* d_ws, size_t ws_size,
                              hipStream_t stream) {
  const float* x      = (const float*)d_in[0];   // [B,T,C]
  const float* w_attn = (const float*)d_in[1];   // [3C,C]
  const float* w_proj = (const float*)d_in[2];   // [C,C]
  float* out = (float*)d_out;                    // [B,T,C]

  const int C = NHEAD * HDIM;                    // 1024
  const int B = 2;
  const int T = in_sizes[0] / (B * C);           // 2048
  const int M = B * T;                           // 4096

  // workspace layout (all bf16/ushort)
  unsigned short* xb  = (unsigned short*)d_ws;            // [M, C]
  unsigned short* wab = xb  + (size_t)M * C;              // [3C, C]
  unsigned short* wpb = wab + (size_t)3 * C * C;          // [C, C]
  unsigned short* qkv = wpb + (size_t)C * C;              // [M, 3C]
  unsigned short* yb  = qkv + (size_t)M * 3 * C;          // [M, C]
  unsigned short* vt  = yb  + (size_t)M * C;              // [B, C, T]

  dim3 blk(256);

  // 0) fp32 -> bf16 conversions
  int n8x = (M * C) / 8, n8a = (3 * C * C) / 8, n8p = (C * C) / 8;
  cvt_f32_bf16<<<dim3((n8x + 255) / 256), blk, 0, stream>>>(x, xb, n8x);
  cvt_f32_bf16<<<dim3((n8a + 255) / 256), blk, 0, stream>>>(w_attn, wab, n8a);
  cvt_f32_bf16<<<dim3((n8p + 255) / 256), blk, 0, stream>>>(w_proj, wpb, n8p);

  // 1) qkv = x @ w_attn^T   (block = 256x64 tile, async-staged weights)
  gemm_bb<false><<<dim3((M / 256) * ((3 * C) / 64)), blk, 0, stream>>>(
      xb, wab, (void*)qkv, M, 3 * C, C);

  // 2) V transpose for key-contiguous P*V loads
  int ne = M * C;
  vtrans_kernel<<<dim3((ne + 255) / 256), blk, 0, stream>>>(qkv, vt, B, T);

  // 3) masked flash attention -> y (bf16)
  int wavesA = B * NHEAD * (T / 16);
  attn_kernel<<<dim3((wavesA + 7) / 8), blk, 0, stream>>>(qkv, vt, yb, B, T);

  // 4) out = y @ w_proj^T
  gemm_bb<true><<<dim3((M / 256) * (C / 64)), blk, 0, stream>>>(
      yb, wpb, (void*)out, M, C, C);
}
